// SlidingWindowAttention_16320875725292
// MI455X (gfx1250) — matrix operand order repaired and checked by reading, not compiled, not run on hardware
//
#include <hip/hip_runtime.h>
#include <hip/hip_bf16.h>

// Problem dims (fixed by the reference)
#define L_SEQ   2048
#define D_MODEL 1024
#define N_QKV   3072
#define N_HEADS 16
#define HD      64
#define WINDOW  256

typedef __attribute__((ext_vector_type(16))) __bf16 v16bf;
typedef __attribute__((ext_vector_type(8)))  float  v8f;

union Frag { uint4 u[2]; v16bf v; };

__device__ __forceinline__ unsigned short f2bf(float f) {
  unsigned u = __builtin_bit_cast(unsigned, f);
  u += 0x7FFFu + ((u >> 16) & 1u);          // round-to-nearest-even
  return (unsigned short)(u >> 16);
}

#define NEG_INF (-__builtin_inff())

// ---------------------------------------------------------------------------
// f32 -> bf16 elementwise convert
__global__ void cvt_bf16_kernel(const float* __restrict__ in,
                                unsigned short* __restrict__ out, int n) {
  int i = blockIdx.x * blockDim.x + threadIdx.x;
  if (i < n) out[i] = f2bf(in[i]);
}

// f32 [rows][cols] -> bf16 transposed [cols][rows]; LDS-tiled so both the
// global reads and writes are coalesced. block (32,8), grid (cols/32, rows/32).
__global__ void cvtT_bf16_kernel(const float* __restrict__ in,
                                 unsigned short* __restrict__ out,
                                 int rows, int cols) {
  __shared__ float tile[32][33];
  const int bx = blockIdx.x * 32;   // col base
  const int by = blockIdx.y * 32;   // row base
  const int tx = threadIdx.x;       // 0..31
  const int ty = threadIdx.y;       // 0..7
#pragma unroll
  for (int i = 0; i < 32; i += 8)
    tile[ty + i][tx] = in[(size_t)(by + ty + i) * cols + bx + tx];
  __syncthreads();
#pragma unroll
  for (int i = 0; i < 32; i += 8)
    out[(size_t)(bx + ty + i) * rows + by + tx] = f2bf(tile[tx][ty + i]);
}

// vt[h][d][l] = qkv_bf16[l][2*D + h*64 + d]  (V transposed per head)
__global__ void build_vt_kernel(const unsigned short* __restrict__ qkvb,
                                unsigned short* __restrict__ vt) {
  int i = blockIdx.x * blockDim.x + threadIdx.x;      // over 16*64*2048
  int l  = i & (L_SEQ - 1);
  int hd = i >> 11;            // h*64 + d
  vt[i] = qkvb[(size_t)l * N_QKV + 2 * D_MODEL + hd];
}

// ---------------------------------------------------------------------------
// One wave per 32x64 output tile (2 A fragments x 4 B fragments -> 8 WMMAs
// per k-step; every B fragment feeds two WMMAs). A: bf16 [M][K] row-major.
// BT: bf16 [N][K] so both fragment loads are contiguous and match the
// v_wmma_f32_16x16x32_bf16 VGPR layouts.
template <bool OUT_F32>
__global__ void gemm_bf16_wmma_kernel(const unsigned short* __restrict__ A,
                                      const unsigned short* __restrict__ BT,
                                      const float* __restrict__ bias,
                                      void* __restrict__ outp,
                                      int M, int N, int K) {
  const int lane = threadIdx.x;           // 0..31, wave32
  const int lo   = lane & 15;
  const int half = lane >> 4;
  const int m0 = blockIdx.x * 32;
  const int n0 = blockIdx.y * 64;

  v8f acc[2][4] = {};
  for (int kk = 0; kk < K; kk += 32) {
    // Two A fragments (rows m0+lo and m0+16+lo); K = half*8 + {0..7,16..23}
    Frag fa0, fa1;
    const unsigned short* pa0 = A + (size_t)(m0 + lo) * K + kk + half * 8;
    fa0.u[0] = *(const uint4*)pa0;
    fa0.u[1] = *(const uint4*)(pa0 + 16);
    const unsigned short* pa1 = pa0 + (size_t)16 * K;
    fa1.u[0] = *(const uint4*)pa1;
    fa1.u[1] = *(const uint4*)(pa1 + 16);
#pragma unroll
    for (int nc = 0; nc < 4; ++nc) {
      // B fragment: col n0+nc*16+lo, K = half*8 + {0..7,16..23}
      Frag fb;
      const unsigned short* pb =
          BT + (size_t)(n0 + nc * 16 + lo) * K + kk + half * 8;
      fb.u[0] = ((const uint4*)pb)[0];
      fb.u[1] = *(const uint4*)(pb + 16);
      acc[0][nc] = __builtin_amdgcn_wmma_f32_16x16x32_bf16(
          false, fa0.v, false, fb.v, (short)0, acc[0][nc], false, false);
      acc[1][nc] = __builtin_amdgcn_wmma_f32_16x16x32_bf16(
          false, fa1.v, false, fb.v, (short)0, acc[1][nc], false, false);
    }
  }

  const int rb = half * 8;
#pragma unroll
  for (int g = 0; g < 2; ++g) {
#pragma unroll
    for (int nc = 0; nc < 4; ++nc) {
      const int col = n0 + nc * 16 + lo;
      const float b = bias[col];
#pragma unroll
      for (int r = 0; r < 8; ++r) {
        const float v = acc[g][nc][r] + b;
        const size_t idx = (size_t)(m0 + g * 16 + rb + r) * N + col;
        if (OUT_F32) ((float*)outp)[idx] = v;
        else         ((unsigned short*)outp)[idx] = f2bf(v);
      }
    }
  }
}

// ---------------------------------------------------------------------------
// Flash-style attention, one wave per (head, 16-query tile).
// qkvb: bf16 [L][3*D] (q at +0, k at +D).  vt: bf16 [h][64][L].
// Mask (reference quirk): attention FORBIDDEN where (j<=i && j>=i-255).
__global__ void attn_wmma_kernel(const unsigned short* __restrict__ qkvb,
                                 const unsigned short* __restrict__ vt,
                                 unsigned short* __restrict__ attnb) {
  __shared__ unsigned short qTile[16 * 64];   // 2 KB
  __shared__ unsigned short pTile[16 * 32];   // 1 KB

  const int lane = threadIdx.x;
  const int lo   = lane & 15;
  const int half = lane >> 4;
  const int rb   = half * 8;
  const int h  = blockIdx.y;
  const int q0 = blockIdx.x * 16;

  // ---- async copy of the Q tile (16 rows x 64 bf16 = 2048 B) into LDS ----
  {
    const unsigned ldsBase = (unsigned)(size_t)&qTile[0];
    const char* gbase = (const char*)qkvb;
#pragma unroll
    for (int t = 0; t < 4; ++t) {
      const unsigned boff = (unsigned)(t * 512 + lane * 16);
      const unsigned row = boff >> 7;       // 128 B per Q row
      const unsigned col = boff & 127;
      const unsigned long long ga = (unsigned long long)(
          gbase + ((size_t)(q0 + row) * N_QKV + h * HD) * 2 + col);
      const unsigned laddr = ldsBase + boff;
      asm volatile("global_load_async_to_lds_b128 %0, %1, off"
                   :: "v"(laddr), "v"(ga) : "memory");
    }
    asm volatile("s_wait_asynccnt 0" ::: "memory");
  }
  __syncthreads();

  // Q fragments for the two 32-wide d-steps
  Frag qa[2];
#pragma unroll
  for (int ks = 0; ks < 2; ++ks) {
    const unsigned short* p = qTile + lo * 64 + ks * 32 + half * 16;
    qa[ks].u[0] = *(const uint4*)p;
    qa[ks].u[1] = *(const uint4*)(p + 8);
  }

  float mrun[8], lrun[8];
#pragma unroll
  for (int r = 0; r < 8; ++r) { mrun[r] = NEG_INF; lrun[r] = 0.f; }
  v8f o[4] = {};
  const float scale = 0.125f;   // 1/sqrt(64)

  for (int k0 = 0; k0 < L_SEQ; k0 += 32) {
    // Prefetch next iteration's K rows while this tile computes
    {
      const int kn = (k0 + 32 < L_SEQ) ? (k0 + 32) : k0;
      __builtin_prefetch(qkvb + (size_t)(kn + lo) * N_QKV + D_MODEL + h * HD,
                         0, 3);
      __builtin_prefetch(
          qkvb + (size_t)(kn + 16 + lo) * N_QKV + D_MODEL + h * HD, 0, 3);
    }

    // ---- scores for two 16-key subtiles: S = Q * K^T ----
    v8f s0 = {}, s1 = {};
#pragma unroll
    for (int ks = 0; ks < 2; ++ks) {
      Frag kb0, kb1;
      const unsigned short* pk0 = qkvb + (size_t)(k0 + lo) * N_QKV +
                                  D_MODEL + h * HD + ks * 32 + half * 16;
      kb0.u[0] = ((const uint4*)pk0)[0];
      kb0.u[1] = ((const uint4*)pk0)[1];
      s0 = __builtin_amdgcn_wmma_f32_16x16x32_bf16(
          false, qa[ks].v, false, kb0.v, (short)0, s0, false, false);
      const unsigned short* pk1 = qkvb + (size_t)(k0 + 16 + lo) * N_QKV +
                                  D_MODEL + h * HD + ks * 32 + half * 16;
      kb1.u[0] = ((const uint4*)pk1)[0];
      kb1.u[1] = ((const uint4*)pk1)[1];
      s1 = __builtin_amdgcn_wmma_f32_16x16x32_bf16(
          false, qa[ks].v, false, kb1.v, (short)0, s1, false, false);
    }

    // ---- mask + online softmax (rows live across 16-lane halves) ----
    const int j0 = k0 + lo, j1 = k0 + 16 + lo;
#pragma unroll
    for (int r = 0; r < 8; ++r) {
      const int i = q0 + rb + r;
      float a0 = s0[r] * scale;
      float a1 = s1[r] * scale;
      if ((j0 <= i) && (j0 >= i - (WINDOW - 1))) a0 = NEG_INF;
      if ((j1 <= i) && (j1 >= i - (WINDOW - 1))) a1 = NEG_INF;

      float t = fmaxf(a0, a1);
      t = fmaxf(t, __shfl_xor(t, 1, 32));
      t = fmaxf(t, __shfl_xor(t, 2, 32));
      t = fmaxf(t, __shfl_xor(t, 4, 32));
      t = fmaxf(t, __shfl_xor(t, 8, 32));
      const float mn = fmaxf(mrun[r], t);

      float corr, p0, p1;
      if (mn == NEG_INF) { corr = 1.f; p0 = 0.f; p1 = 0.f; }
      else {
        corr = (mrun[r] == NEG_INF) ? 0.f : __expf(mrun[r] - mn);
        p0 = __expf(a0 - mn);     // expf(-inf) == 0 handles masked entries
        p1 = __expf(a1 - mn);
      }
      float ts = p0 + p1;
      ts += __shfl_xor(ts, 1, 32);
      ts += __shfl_xor(ts, 2, 32);
      ts += __shfl_xor(ts, 4, 32);
      ts += __shfl_xor(ts, 8, 32);

      lrun[r] = lrun[r] * corr + ts;
      mrun[r] = mn;
#pragma unroll
      for (int dc = 0; dc < 4; ++dc) o[dc][r] *= corr;

      pTile[(rb + r) * 32 + lo]      = f2bf(p0);
      pTile[(rb + r) * 32 + 16 + lo] = f2bf(p1);
    }
    __syncthreads();

    // ---- O += P * V ----
    Frag pa;
    const unsigned short* pp = pTile + lo * 32 + half * 16;
    pa.u[0] = *(const uint4*)pp;
    pa.u[1] = *(const uint4*)(pp + 8);
#pragma unroll
    for (int dc = 0; dc < 4; ++dc) {
      Frag vb;
      const unsigned short* pv =
          vt + ((size_t)h * HD + dc * 16 + lo) * L_SEQ + k0 + half * 16;
      vb.u[0] = ((const uint4*)pv)[0];
      vb.u[1] = ((const uint4*)pv)[1];
      o[dc] = __builtin_amdgcn_wmma_f32_16x16x32_bf16(
          false, pa.v, false, vb.v, (short)0, o[dc], false, false);
    }
    __syncthreads();
  }

  // ---- normalize and store per-head output (bf16 for the proj GEMM) ----
#pragma unroll
  for (int r = 0; r < 8; ++r) {
    const float inv = 1.0f / fmaxf(lrun[r], 1e-30f);
#pragma unroll
    for (int dc = 0; dc < 4; ++dc) {
      attnb[(size_t)(q0 + rb + r) * D_MODEL + h * HD + dc * 16 + lo] =
          f2bf(o[dc][r] * inv);
    }
  }
}

// ---------------------------------------------------------------------------
extern "C" void kernel_launch(void* const* d_in, const int* in_sizes, int n_in,
                              void* d_out, int out_size, void* d_ws,
                              size_t ws_size, hipStream_t stream) {
  const float* x     = (const float*)d_in[0];   // [2048,1024]
  const float* Wqkv  = (const float*)d_in[1];   // [1024,3072]
  const float* bqkv  = (const float*)d_in[2];   // [3072]
  const float* Wproj = (const float*)d_in[3];   // [1024,1024]
  const float* bproj = (const float*)d_in[4];   // [1024]
  float* out = (float*)d_out;                   // [2048,1024]

  char* ws = (char*)d_ws;
  unsigned short* xb     = (unsigned short*)(ws);                       // 4 MB
  unsigned short* WqkvT  = (unsigned short*)(ws + (4u << 20));          // 6 MB
  unsigned short* WprojT = (unsigned short*)(ws + (10u << 20));         // 2 MB
  unsigned short* qkvb   = (unsigned short*)(ws + (12u << 20));         // 12 MB
  unsigned short* vt     = (unsigned short*)(ws + (24u << 20));         // 4 MB
  unsigned short* attnb  = (unsigned short*)(ws + (28u << 20));         // 4 MB

  // 1) bf16 conversions (weights transposed for contiguous B fragments)
  {
    int n = L_SEQ * D_MODEL;
    cvt_bf16_kernel<<<(n + 255) / 256, 256, 0, stream>>>(x, xb, n);
    cvtT_bf16_kernel<<<dim3(N_QKV / 32, D_MODEL / 32), dim3(32, 8), 0,
                       stream>>>(Wqkv, WqkvT, D_MODEL, N_QKV);
    cvtT_bf16_kernel<<<dim3(D_MODEL / 32, D_MODEL / 32), dim3(32, 8), 0,
                       stream>>>(Wproj, WprojT, D_MODEL, D_MODEL);
  }

  // 2) QKV GEMM: qkvb = bf16(x @ Wqkv + bqkv)
  gemm_bf16_wmma_kernel<false>
      <<<dim3(L_SEQ / 32, N_QKV / 64), 32, 0, stream>>>(
          xb, WqkvT, bqkv, qkvb, L_SEQ, N_QKV, D_MODEL);

  // 3) transpose V per head for contiguous PV B-fragments
  {
    int n = N_HEADS * HD * L_SEQ;
    build_vt_kernel<<<(n + 255) / 256, 256, 0, stream>>>(qkvb, vt);
  }

  // 4) attention
  attn_wmma_kernel<<<dim3(L_SEQ / 16, N_HEADS), 32, 0, stream>>>(qkvb, vt,
                                                                 attnb);

  // 5) projection: out = attn @ Wproj + bproj (f32 output)
  gemm_bf16_wmma_kernel<true>
      <<<dim3(L_SEQ / 32, D_MODEL / 64), 32, 0, stream>>>(
          attnb, WprojT, bproj, out, L_SEQ, D_MODEL, D_MODEL);
}
